// CausalMultiHeadSelfAttention_36481452212426
// MI455X (gfx1250) — matrix operand order repaired and checked by reading, not compiled, not run on hardware
//
#include <hip/hip_runtime.h>
#include <hip/hip_bf16.h>

// ---------------------------------------------------------------------------
// CDNA5 (gfx1250, wave32) causal MHA with RoPE, bf16 WMMA + f32 accumulate.
// ---------------------------------------------------------------------------

typedef __attribute__((ext_vector_type(16))) __bf16 v16bf;
typedef __attribute__((ext_vector_type(8)))  __bf16 v8bf;
typedef __attribute__((ext_vector_type(4)))  __bf16 v4bf;
typedef __attribute__((ext_vector_type(8)))  float  v8f;

#define D_MODEL 1024
#define SEQ     2048
#define BATCH   4
#define NH      16
#define DK      64
// ln(10000)/32  (theta^(-2i/64) = exp(-i * ln(theta)/32))
#define ROPE_LNF 0.2878231366f

static __device__ __forceinline__ v8f vzero8() {
    v8f z;
#pragma unroll
    for (int i = 0; i < 8; ++i) z[i] = 0.0f;
    return z;
}

static __device__ __forceinline__ v8f wmma_bf16(v16bf a, v16bf b, v8f c) {
    // D = A(16x32 bf16) * B(32x16 bf16) + C(16x16 f32)
    return __builtin_amdgcn_wmma_f32_16x16x32_bf16(
        /*neg_a=*/false, a, /*neg_b=*/false, b,
        /*c_mod=*/(short)0, c, /*reuse_a=*/false, /*reuse_b=*/false);
}

// A-matrix 16x32 bf16 fragment (ISA 7.12.2):
//  lanes 0-15  hold row M=lane,    K slots {k0+0..7,  k0+16..23}
//  lanes 16-31 hold row M=lane-16, K slots {k0+8..15, k0+24..31}
// `row` = pointer to this lane's matrix row; `half` = lane>>4.
static __device__ __forceinline__ v16bf load_afrag(const __bf16* row, int k0, int half) {
    const v8bf lo = *(const v8bf*)(row + k0 + half * 8);
    const v8bf hi = *(const v8bf*)(row + k0 + 16 + half * 8);
    v16bf a;
#pragma unroll
    for (int i = 0; i < 8; ++i) { a[i] = lo[i]; a[8 + i] = hi[i]; }
    return a;
}

// B-matrix 32x16 bf16 fragment: lane n holds column N=n%16;
// lanes 0-15 hold K slots {k0+0..7, k0+16..23}, lanes 16-31 hold {k0+8..15, k0+24..31}.
// `col` = pointer to the memory row representing column N (transposed operand).
static __device__ __forceinline__ v16bf load_bfrag(const __bf16* col, int k0, int half) {
    const v8bf lo = *(const v8bf*)(col + k0 + half * 8);
    const v8bf hi = *(const v8bf*)(col + k0 + 16 + half * 8);
    v16bf b;
#pragma unroll
    for (int i = 0; i < 8; ++i) { b[i] = lo[i]; b[8 + i] = hi[i]; }
    return b;
}

// ---------------------------------------------------------------------------
// fp32 -> bf16 bulk conversion (vectorized, grid-strided)
// ---------------------------------------------------------------------------
__global__ void cvt_bf16_kernel(const float* __restrict__ src,
                                __bf16* __restrict__ dst, int n4) {
    int i = blockIdx.x * blockDim.x + threadIdx.x;
    const int stride = gridDim.x * blockDim.x;
    for (; i < n4; i += stride) {
        const float4 v = ((const float4*)src)[i];
        v4bf o;
        o[0] = (__bf16)v.x; o[1] = (__bf16)v.y;
        o[2] = (__bf16)v.z; o[3] = (__bf16)v.w;
        ((v4bf*)dst)[i] = o;
    }
}

// ---------------------------------------------------------------------------
// Projection GEMM: Y = X @ W^T  (8192x1024 @ 1024x1024), optional RoPE on the
// fp32 accumulators, output scattered per head as bf16.
//   transposed==0 : out[b][h][s][dk]   (Q, K)
//   transposed==1 : out[b][h][dk][s]   (V, so PV B-fragments are contiguous)
// grid (D_MODEL/64, B*S/64), block 128 (4 waves, each wave: 16 rows x 64 cols)
// ---------------------------------------------------------------------------
__global__ __launch_bounds__(128)
void proj_rope_kernel(const __bf16* __restrict__ X, const __bf16* __restrict__ W,
                      __bf16* __restrict__ out, int do_rope, int transposed) {
    const int wave   = threadIdx.x >> 5;
    const int lane   = threadIdx.x & 31;
    const int lane16 = lane & 15;
    const int half   = lane >> 4;

    const int Cb = blockIdx.x * 64;              // 64 cols == one head
    const int Rb = blockIdx.y * 64 + wave * 16;  // 16 rows per wave
    const int hidx = Cb >> 6;

    const __bf16* xrow = X + (size_t)(Rb + lane16) * D_MODEL;

    v8f acc[4];
#pragma unroll
    for (int c = 0; c < 4; ++c) acc[c] = vzero8();

    for (int k = 0; k < D_MODEL; k += 32) {
        // WGP-scope prefetch (locality 3) -> fills all cache levels
        __builtin_prefetch(xrow + k + 128, 0, 3);   // global_prefetch_b8
        const v16bf a = load_afrag(xrow, k, half);
#pragma unroll
        for (int c = 0; c < 4; ++c) {
            const __bf16* wrow = W + (size_t)(Cb + c * 16 + lane16) * D_MODEL;
            const v16bf b = load_bfrag(wrow, k, half);
            acc[c] = wmma_bf16(a, b, acc[c]);
        }
    }

    if (do_rope) {
        // C/D layout: element (vgpr r, lane): row M = r + 8*half, col N = lane16.
        // Adjacent cols (2i, 2i+1) are adjacent lanes -> pair via shfl_xor(.,1).
#pragma unroll
        for (int c = 0; c < 4; ++c) {
            const int dim  = c * 16 + lane16;          // 0..63 within head
            const float invf = __expf(-ROPE_LNF * (float)(dim >> 1));
#pragma unroll
            for (int r = 0; r < 8; ++r) {
                const int gr  = Rb + r + 8 * half;
                const float pos = (float)(gr & (SEQ - 1));
                const float ang = pos * invf;
                const float cs = __cosf(ang), sn = __sinf(ang);
                const float x  = acc[c][r];
                const float xp = __shfl_xor(x, 1, 32); // partner (other of pair)
                acc[c][r] = (dim & 1) ? (xp * sn + x * cs)   // odd  : x1*sin + x2*cos
                                      : (x * cs - xp * sn);  // even : x1*cos - x2*sin
            }
        }
    }

#pragma unroll
    for (int c = 0; c < 4; ++c) {
        const int dim = c * 16 + lane16;
#pragma unroll
        for (int r = 0; r < 8; ++r) {
            const int gr   = Rb + r + 8 * half;
            const int bb   = gr >> 11;        // / SEQ
            const int sidx = gr & (SEQ - 1);
            size_t idx;
            if (transposed)
                idx = (((size_t)(bb * NH + hidx) * DK + dim) * SEQ) + sidx;
            else
                idx = (((size_t)(bb * NH + hidx) * SEQ + sidx) * DK) + dim;
            out[idx] = (__bf16)acc[c][r];
        }
    }
}

// ---------------------------------------------------------------------------
// Flash attention (causal, online softmax).
// grid (SEQ/64, NH, BATCH), block 128. Each wave: 16 q-rows, full dk=64.
// Per 32-key block: 4 WMMAs for S = Q K^T, 4 WMMAs for O += P V.
// ---------------------------------------------------------------------------
__global__ __launch_bounds__(128)
void flash_attn_kernel(const __bf16* __restrict__ Q, const __bf16* __restrict__ K,
                       const __bf16* __restrict__ Vt, __bf16* __restrict__ AO) {
    const int wave   = threadIdx.x >> 5;
    const int lane   = threadIdx.x & 31;
    const int lane16 = lane & 15;
    const int half   = lane >> 4;

    const int b  = blockIdx.z;
    const int h  = blockIdx.y;
    const int qb = blockIdx.x * 64 + wave * 16;

    const __bf16* Qh = Q  + (size_t)(b * NH + h) * SEQ * DK;
    const __bf16* Kh = K  + (size_t)(b * NH + h) * SEQ * DK;
    const __bf16* Vh = Vt + (size_t)(b * NH + h) * DK * SEQ;

    // Per-wave P staging: D-layout -> A-fragment layout via LDS (pad to 40).
    __shared__ __align__(16) __bf16 Plds[4][16][40];

    // Q A-fragments for both d_k halves (reused across all key blocks).
    const __bf16* qrow = Qh + (size_t)(qb + lane16) * DK;
    v16bf qf[2];
    qf[0] = load_afrag(qrow, 0, half);
    qf[1] = load_afrag(qrow, 32, half);

    float mrow[8], lrow[8];
    v8f o[4];
#pragma unroll
    for (int r = 0; r < 8; ++r) { mrow[r] = -__builtin_inff(); lrow[r] = 0.0f; }
#pragma unroll
    for (int c = 0; c < 4; ++c) o[c] = vzero8();

    const int nkb = (qb + 15) / 32 + 1;   // causal: keys up to qb+15
    for (int kb = 0; kb < nkb; ++kb) {
        const int key0 = kb * 32;
        __builtin_prefetch(Kh + (size_t)(key0 + 32 + lane16) * DK, 0, 3);

        // ---- scores: S(16x32) = Q(16x64) . K^T, two 16-key subtiles --------
        v8f s[2];
#pragma unroll
        for (int n = 0; n < 2; ++n) {
            const __bf16* krow = Kh + (size_t)(key0 + n * 16 + lane16) * DK;
            v8f z = vzero8();
            z = wmma_bf16(qf[0], load_bfrag(krow, 0,  half), z);
            z = wmma_bf16(qf[1], load_bfrag(krow, 32, half), z);
            s[n] = z;
        }

        // ---- scale + causal mask ------------------------------------------
#pragma unroll
        for (int n = 0; n < 2; ++n)
#pragma unroll
            for (int r = 0; r < 8; ++r) {
                const int qr  = qb + r + 8 * half;
                const int key = key0 + n * 16 + lane16;
                const float v = s[n][r] * 0.125f;     // 1/sqrt(64)
                s[n][r] = (key <= qr) ? v : -__builtin_inff();
            }

        // ---- online softmax (rows live across 16 lanes of each half) ------
#pragma unroll
        for (int r = 0; r < 8; ++r) {
            float t = fmaxf(s[0][r], s[1][r]);
            t = fmaxf(t, __shfl_xor(t, 1, 32));
            t = fmaxf(t, __shfl_xor(t, 2, 32));
            t = fmaxf(t, __shfl_xor(t, 4, 32));
            t = fmaxf(t, __shfl_xor(t, 8, 32));
            const float newm  = fmaxf(mrow[r], t);
            const float alpha = __expf(mrow[r] - newm);
            mrow[r] = newm;
            const float p0 = __expf(s[0][r] - newm);
            const float p1 = __expf(s[1][r] - newm);
            s[0][r] = p0; s[1][r] = p1;
            float rs = p0 + p1;
            rs += __shfl_xor(rs, 1, 32);
            rs += __shfl_xor(rs, 2, 32);
            rs += __shfl_xor(rs, 4, 32);
            rs += __shfl_xor(rs, 8, 32);
            lrow[r] = lrow[r] * alpha + rs;
#pragma unroll
            for (int c = 0; c < 4; ++c) o[c][r] *= alpha;
        }

        // ---- P: D-layout -> A-fragment layout through LDS -----------------
#pragma unroll
        for (int n = 0; n < 2; ++n)
#pragma unroll
            for (int r = 0; r < 8; ++r)
                Plds[wave][r + 8 * half][n * 16 + lane16] = (__bf16)s[n][r];
        asm volatile("s_wait_dscnt 0" ::: "memory");   // wave-local RAW on LDS

        const v16bf pf = load_afrag(&Plds[wave][lane16][0], 0, half);

        // ---- O(16x64) += P(16x32) . V(32x64); V stored transposed ---------
#pragma unroll
        for (int c = 0; c < 4; ++c) {
            const __bf16* vcol = Vh + (size_t)(c * 16 + lane16) * SEQ;
            o[c] = wmma_bf16(pf, load_bfrag(vcol, key0, half), o[c]);
        }
    }

    // ---- normalize + store bf16 attention output [b][s][h*64+dim] ---------
#pragma unroll
    for (int c = 0; c < 4; ++c) {
        const int dim = c * 16 + lane16;
#pragma unroll
        for (int r = 0; r < 8; ++r) {
            const int sidx = qb + r + 8 * half;
            const float v  = o[c][r] / lrow[r];
            AO[((size_t)b * SEQ + sidx) * D_MODEL + h * DK + dim] = (__bf16)v;
        }
    }
}

// ---------------------------------------------------------------------------
// Output projection: out(f32) = AO(bf16) @ Wo^T(bf16)
// ---------------------------------------------------------------------------
__global__ __launch_bounds__(128)
void out_proj_kernel(const __bf16* __restrict__ AO, const __bf16* __restrict__ Wo,
                     float* __restrict__ out) {
    const int wave   = threadIdx.x >> 5;
    const int lane   = threadIdx.x & 31;
    const int lane16 = lane & 15;
    const int half   = lane >> 4;

    const int Cb = blockIdx.x * 64;
    const int Rb = blockIdx.y * 64 + wave * 16;

    const __bf16* arow = AO + (size_t)(Rb + lane16) * D_MODEL;

    v8f acc[4];
#pragma unroll
    for (int c = 0; c < 4; ++c) acc[c] = vzero8();

    for (int k = 0; k < D_MODEL; k += 32) {
        __builtin_prefetch(arow + k + 128, 0, 3);
        const v16bf a = load_afrag(arow, k, half);
#pragma unroll
        for (int c = 0; c < 4; ++c) {
            const __bf16* wrow = Wo + (size_t)(Cb + c * 16 + lane16) * D_MODEL;
            acc[c] = wmma_bf16(a, load_bfrag(wrow, k, half), acc[c]);
        }
    }

#pragma unroll
    for (int c = 0; c < 4; ++c) {
        const int col = Cb + c * 16 + lane16;
#pragma unroll
        for (int r = 0; r < 8; ++r) {
            const int gr = Rb + r + 8 * half;
            out[(size_t)gr * D_MODEL + col] = acc[c][r];
        }
    }
}

// ---------------------------------------------------------------------------
// Host launch
// ---------------------------------------------------------------------------
extern "C" void kernel_launch(void* const* d_in, const int* in_sizes, int n_in,
                              void* d_out, int out_size, void* d_ws, size_t ws_size,
                              hipStream_t stream) {
    (void)in_sizes; (void)n_in; (void)out_size; (void)ws_size;

    const float* X  = (const float*)d_in[0];
    const float* Wq = (const float*)d_in[1];
    const float* Wk = (const float*)d_in[2];
    const float* Wv = (const float*)d_in[3];
    const float* Wo = (const float*)d_in[4];
    // d_in[5] = token_positions (arange(SEQ)) — computed implicitly.
    float* out = (float*)d_out;

    char* ws = (char*)d_ws;
    const size_t NX = (size_t)BATCH * SEQ * D_MODEL;   // 8 Mi elems
    const size_t NW = (size_t)D_MODEL * D_MODEL;       // 1 Mi elems
    __bf16* Xbf  = (__bf16*)(ws);
    __bf16* Wqbf = (__bf16*)(ws + 2 * NX);
    __bf16* Wkbf = (__bf16*)(ws + 2 * (NX + NW));
    __bf16* Wvbf = (__bf16*)(ws + 2 * (NX + 2 * NW));
    __bf16* Wobf = (__bf16*)(ws + 2 * (NX + 3 * NW));
    __bf16* Qbf  = (__bf16*)(ws + 2 * (NX + 4 * NW));
    __bf16* Kbf  = (__bf16*)(ws + 2 * (2 * NX + 4 * NW));
    __bf16* Vtbf = (__bf16*)(ws + 2 * (3 * NX + 4 * NW));
    __bf16* AObf = (__bf16*)(ws + 2 * (4 * NX + 4 * NW));

    // 1) fp32 -> bf16 conversions
    cvt_bf16_kernel<<<4096, 256, 0, stream>>>(X,  Xbf,  (int)(NX / 4));
    cvt_bf16_kernel<<<1024, 256, 0, stream>>>(Wq, Wqbf, (int)(NW / 4));
    cvt_bf16_kernel<<<1024, 256, 0, stream>>>(Wk, Wkbf, (int)(NW / 4));
    cvt_bf16_kernel<<<1024, 256, 0, stream>>>(Wv, Wvbf, (int)(NW / 4));
    cvt_bf16_kernel<<<1024, 256, 0, stream>>>(Wo, Wobf, (int)(NW / 4));

    // 2) projections (+RoPE for Q,K; V stored transposed per head)
    const dim3 pg(D_MODEL / 64, (BATCH * SEQ) / 64);
    proj_rope_kernel<<<pg, 128, 0, stream>>>(Xbf, Wqbf, Qbf,  1, 0);
    proj_rope_kernel<<<pg, 128, 0, stream>>>(Xbf, Wkbf, Kbf,  1, 0);
    proj_rope_kernel<<<pg, 128, 0, stream>>>(Xbf, Wvbf, Vtbf, 0, 1);

    // 3) causal flash attention
    flash_attn_kernel<<<dim3(SEQ / 64, NH, BATCH), 128, 0, stream>>>(Qbf, Kbf, Vtbf, AObf);

    // 4) output projection -> fp32
    out_proj_kernel<<<pg, 128, 0, stream>>>(AObf, Wobf, out);
}
